// LSTM_25305947308140
// MI455X (gfx1250) — matrix operand order repaired and checked by reading, not compiled, not run on hardware
//
#include <hip/hip_runtime.h>
#include <hip/hip_bf16.h>

typedef __attribute__((ext_vector_type(16))) _Float16 v16h;
typedef __attribute__((ext_vector_type(8)))  float    v8f;

#define WMMA_F32_F16(A, B, C) \
  __builtin_amdgcn_wmma_f32_16x16x32_f16(false, (A), false, (B), (short)0, (C), false, false)

// CDNA5 has V_TANH_F32; use it when the builtin is exposed.
#if __has_builtin(__builtin_amdgcn_tanhf)
__device__ __forceinline__ float ftanh_fast(float x) { return __builtin_amdgcn_tanhf(x); }
#else
__device__ __forceinline__ float ftanh_fast(float x) {
  float e = __expf(-2.0f * x);
  return (1.0f - e) / (1.0f + e);
}
#endif
// sigmoid(x) = 0.5*tanh(x/2) + 0.5  -> mul + v_tanh + fma
__device__ __forceinline__ float fsigmoid(float x) {
  return fmaf(0.5f, ftanh_fast(0.5f * x), 0.5f);
}

// Fully fused LSTM: each workgroup owns 16 batch rows for all T steps.
// 8 waves; wave w owns h-columns [w*16, w*16+16) across all 4 gates, so the
// gate mix is wave-local and c-state lives in a v8f accumulator register.
__global__ __launch_bounds__(256, 1)
void lstm_fused_kernel(const float* __restrict__ x,      // [B, T, F]
                       const float* __restrict__ W_ih,   // [4H, F]
                       const float* __restrict__ W_hh,   // [4H, H]
                       const float* __restrict__ b_ih,   // [4H]
                       const float* __restrict__ b_hh,   // [4H]
                       const float* __restrict__ W_fc,   // [1, H]
                       const float* __restrict__ b_fc,   // [1]
                       float* __restrict__ out)          // [B, T, 1]
{
  constexpr int T = 128, F = 32, H = 128;

  __shared__ _Float16 h_lds[16 * H];   // h_t tile, row-major [16 rows][128 cols], f16
  __shared__ float outPart[8][16];     // per-wave partial FC dot per row

  const int tid   = threadIdx.x;
  const int wave  = tid >> 5;
  const int lane  = tid & 31;
  const int lhalf = lane & 15;         // 0..15
  const int hi    = lane >> 4;         // 0 or 1 (lane half)
  const int b0    = blockIdx.x * 16;   // batch row base for this workgroup
  const int ncol  = wave * 16 + lhalf; // h column owned by this lane (0..127)

  // ---------------- Resident WMMA B-fragments (live in VGPRs across the T loop) ----
  // B is 32x16 (KxN) f16: K = hi*8 + e (e<8), 16 + hi*8 + (e-8) (e>=8), N = lhalf.
  // B[k][n] = W[gate*H + (colbase + n)][k]  (W row-major [4H, K])
  v16h  Bhh[4][4];   // [gate][kstep]  -> 128 VGPRs
  v16h  Bih[4];      // [gate], K = F = 32 exactly one fragment
  float bias[4];
  #pragma unroll
  for (int g = 0; g < 4; ++g) {
    const float* wr = W_hh + (size_t)(g * H + ncol) * H;
    #pragma unroll
    for (int ks = 0; ks < 4; ++ks) {
      v16h f;
      #pragma unroll
      for (int e = 0; e < 16; ++e) f[e] = (_Float16)wr[ks * 32 + (e < 8 ? hi * 8 + e : 16 + hi * 8 + (e - 8))];
      Bhh[g][ks] = f;
    }
    const float* wi = W_ih + (size_t)(g * H + ncol) * F;
    v16h fi;
    #pragma unroll
    for (int e = 0; e < 16; ++e) fi[e] = (_Float16)wi[(e < 8 ? hi * 8 + e : 16 + hi * 8 + (e - 8))];
    Bih[g]  = fi;
    bias[g] = b_ih[g * H + ncol] + b_hh[g * H + ncol];
  }
  const float wfc = W_fc[ncol];
  const float bfc = b_fc[0];

  // c state lives in the WMMA C/D accumulator layout: v8f, one tile per wave.
  v8f c = {};
  for (int i = tid; i < 16 * H; i += 256) h_lds[i] = (_Float16)0.0f;
  __syncthreads();

  const float* xrow = x + (size_t)(b0 + lhalf) * T * F;  // A-matrix row M = lhalf

  // Double-buffered x row in the 16-bit A-fragment K order:
  //   xr[0..7]  <- xp[hi*8 + 0..7],   xr[8..15] <- xp[16 + hi*8 + 0..7]
  float xr[16];
  {
    const float* xp = xrow;
    #pragma unroll
    for (int e = 0; e < 8; ++e) { xr[e] = xp[hi * 8 + e]; xr[e + 8] = xp[16 + hi * 8 + e]; }
  }

  #pragma unroll 1
  for (int t = 0; t < T; ++t) {
    // ---- convert buffered x row into the f16 A fragment
    v16h Ax;
    #pragma unroll
    for (int e = 0; e < 16; ++e) Ax[e] = (_Float16)xr[e];

    // ---- gate pre-activations: acc = bias + x_t @ W_ih^T  (K = F = 32, one WMMA)
    v8f acc[4];
    #pragma unroll
    for (int g = 0; g < 4; ++g) {
      v8f a;
      #pragma unroll
      for (int r = 0; r < 8; ++r) a[r] = bias[g];
      acc[g] = WMMA_F32_F16(Ax, Bih[g], a);
    }

    // ---- preload ALL recurrent A fragments (8x ds_load_b128 issued together)
    v16h Ah[4];
    #pragma unroll
    for (int ks = 0; ks < 4; ++ks) {
      const _Float16* hp = &h_lds[lhalf * H + ks * 32 + hi * 8];
      #pragma unroll
      for (int e = 0; e < 8; ++e) { Ah[ks][e] = hp[e]; Ah[ks][e + 8] = hp[16 + e]; }
    }

    // ---- fetch next timestep's x row now; global latency hides under 16 WMMAs
    if (t + 1 < T) {
      const float* xn = xrow + (t + 1) * F;
      #pragma unroll
      for (int e = 0; e < 8; ++e) { xr[e] = xn[hi * 8 + e]; xr[e + 8] = xn[16 + hi * 8 + e]; }
      if (t + 2 < T) __builtin_prefetch(xrow + (t + 2) * F, 0, 0);  // warm L2
    }

    // ---- recurrent part: acc += h_{t-1} @ W_hh^T  (4 k-steps x 4 gates)
    #pragma unroll
    for (int ks = 0; ks < 4; ++ks)
      #pragma unroll
      for (int g = 0; g < 4; ++g)
        acc[g] = WMMA_F32_F16(Ah[ks], Bhh[g][ks], acc[g]);

    __syncthreads();   // all waves finished reading h_{t-1} from LDS

    // ---- gate nonlinearities + state update, all in accumulator layout
    v8f hnew;
    #pragma unroll
    for (int r = 0; r < 8; ++r) {
      float ig = fsigmoid(acc[0][r]);
      float fg = fsigmoid(acc[1][r]);
      float gg = ftanh_fast(acc[2][r]);
      float og = fsigmoid(acc[3][r]);
      float cn = fg * c[r] + ig * gg;
      c[r]    = cn;
      hnew[r] = og * ftanh_fast(cn);
    }
    // store h_t tile back to LDS (f16): lane col = ncol, row = r + hi*8
    #pragma unroll
    for (int r = 0; r < 8; ++r)
      h_lds[(r + hi * 8) * H + ncol] = (_Float16)hnew[r];

    // ---- fused FC head: per-row dot over this wave's 16 columns (wave32 shuffles)
    #pragma unroll
    for (int r = 0; r < 8; ++r) {
      float v = hnew[r] * wfc;
      v += __shfl_xor(v, 1, 32);
      v += __shfl_xor(v, 2, 32);
      v += __shfl_xor(v, 4, 32);
      v += __shfl_xor(v, 8, 32);   // reduced within each 16-lane half
      if (lhalf == 0) outPart[wave][hi * 8 + r] = v;
    }
    __syncthreads();   // h_lds (h_t) and outPart valid
    if (wave == 0 && lane < 16) {
      float s = bfc;
      #pragma unroll
      for (int w = 0; w < 8; ++w) s += outPart[w][lane];
      out[(size_t)(b0 + lane) * T + t] = s;
    }
    // Other waves may race ahead into the next iteration's h_lds *reads* (safe:
    // h_t is final) but cannot overwrite h or outPart until the next first
    // barrier, which wave 0 only signals after its outPart reads complete.
  }
}

extern "C" void kernel_launch(void* const* d_in, const int* in_sizes, int n_in,
                              void* d_out, int out_size, void* d_ws, size_t ws_size,
                              hipStream_t stream) {
  const float* x    = (const float*)d_in[0];
  const float* W_ih = (const float*)d_in[1];
  const float* W_hh = (const float*)d_in[2];
  const float* b_ih = (const float*)d_in[3];
  const float* b_hh = (const float*)d_in[4];
  const float* W_fc = (const float*)d_in[5];
  const float* b_fc = (const float*)d_in[6];
  float* out = (float*)d_out;

  const int B = 2048;
  dim3 grid(B / 16), block(256);
  hipLaunchKernelGGL(lstm_fused_kernel, grid, block, 0, stream,
                     x, W_ih, W_hh, b_ih, b_hh, W_fc, b_fc, out);
}
